// Network_27015344291890
// MI455X (gfx1250) — compile-verified, matrix operand order checked by reading, not run on hardware
//
#include <hip/hip_runtime.h>
#include <math.h>

typedef __attribute__((ext_vector_type(16))) _Float16 v16h;
typedef __attribute__((ext_vector_type(8)))  float    v8f;
typedef __attribute__((ext_vector_type(4)))  unsigned int v4u;

#define N_AGENTS 4096
#define E 64
#define FMAP_ROWS 82          // 81 pixels + 1 zero pad row for branchless halo

// ---------------------------------------------------------------------------
// Weight pre-swizzle: (O=128, I=128, 3, 3) f32 -> WMMA B-fragment f16 layout.
// K ordering: k = tap*128 + cin  (tap-major so each 32-K step is one tap).
// Fragment: [conv][ktile(36)][ntile(8)][lane(32)][16 f16], lane&15 = N col,
// K-local mapping mirrors the 16-bit A layout (lanes 0-15: K0-7/16-23,
// lanes 16-31: K8-15/24-31).
// ---------------------------------------------------------------------------
__global__ __launch_bounds__(256) void pack_weights(
    const float* __restrict__ w0, const float* __restrict__ w1,
    const float* __restrict__ w2, const float* __restrict__ w3,
    _Float16* __restrict__ wp)
{
    int idx = blockIdx.x * blockDim.x + threadIdx.x;
    if (idx >= 4 * 36 * 8 * 512) return;
    int conv = idx / 147456;
    int rem  = idx - conv * 147456;
    int kt   = rem / 4096;
    int rem2 = rem - kt * 4096;
    int nt   = rem2 >> 9;
    int e    = rem2 & 511;
    int lane = e >> 4;
    int f    = e & 15;
    int v    = f >> 1, hh = f & 1;
    int ksel = (lane >> 4) * 8;
    int kloc = (v < 4) ? (2 * v + hh + ksel) : (16 + 2 * (v - 4) + hh + ksel);
    int k    = kt * 32 + kloc;
    int tap  = k >> 7;
    int cin  = k & 127;
    int n    = nt * 16 + (lane & 15);
    const float* w = (conv == 0) ? w0 : (conv == 1) ? w1 : (conv == 2) ? w2 : w3;
    wp[idx] = (_Float16)w[(n * 128 + cin) * 9 + tap];
}

// ---------------------------------------------------------------------------
// Fully fused per-agent network: conv1 -> 2x residual conv blocks (WMMA) ->
// conv2 1x1 -> fc_obs -> pos MLP -> cat residual MLP -> GRU -> h out.
// One workgroup (256 threads = 8 waves) per agent; feature maps live in LDS.
// ---------------------------------------------------------------------------
struct NetP {
    const float *obs, *pos, *hid;
    const float *c1w, *c1b;
    const float *r1ab, *r1bb, *r2ab, *r2bb;
    const float *c2w, *c2b, *fcw, *fcb;
    const float *pw, *pb, *p1w, *p1b, *p2w, *p2b;
    const float *cw1, *cb1, *cw2, *cb2;
    const float *wih, *whh, *bih, *bhh;
    const _Float16 *wpack;
    float *hout;
};

__device__ __forceinline__ void conv_wmma(
    const _Float16* __restrict__ xin, _Float16* __restrict__ yout,
    const _Float16* __restrict__ res,               // residual source or null
    const _Float16* __restrict__ wpc,               // this conv's packed B frags
    const float* __restrict__ bias, int lane, int ntile)
{
    const int mrow = lane & 15;
    const int ksel = (lane >> 4) * 8;
    v8f acc[6] = {};
    int prow[6], pcol[6];
    bool pval[6];
#pragma unroll
    for (int m = 0; m < 6; ++m) {
        int p = m * 16 + mrow;
        pval[m] = (p < 81); prow[m] = p / 9; pcol[m] = p % 9;
    }
    const _Float16* bptr = wpc + ntile * 512 + lane * 16;
    // keep the tap loop ROLLED: bounds VGPR pressure (<256, no vgpr-msb thrash)
#pragma unroll 1
    for (int tap = 0; tap < 9; ++tap) {
        const int dh = tap / 3 - 1, dw = tap % 3 - 1;
        // branchless halo: OOB taps read the zero pad row (81)
        int srcoff[6];
#pragma unroll
        for (int m = 0; m < 6; ++m) {
            int hh = prow[m] + dh, ww = pcol[m] + dw;
            bool ok = pval[m] && ((unsigned)hh < 9u) && ((unsigned)ww < 9u);
            int pix = ok ? (hh * 9 + ww) : 81;
            srcoff[m] = pix * 128 + ksel;
        }
#pragma unroll
        for (int c4 = 0; c4 < 4; ++c4) {
            const int c0 = c4 * 32;
            v16h bfrag = *(const v16h*)(bptr + (tap * 4 + c4) * 4096);
            __builtin_prefetch(bptr + (tap * 4 + c4 + 1) * 4096, 0, 1);
            union AU { v16h v; v4u u[2]; } af[6];
#pragma unroll
            for (int m = 0; m < 6; ++m) {          // batch the DS loads
                const _Float16* src = xin + srcoff[m] + c0;
                af[m].u[0] = *(const v4u*)(src);
                af[m].u[1] = *(const v4u*)(src + 16);
            }
#pragma unroll
            for (int m = 0; m < 6; ++m)
                acc[m] = __builtin_amdgcn_wmma_f32_16x16x32_f16(
                    false, af[m].v, false, bfrag, (short)0, acc[m], false, false);
        }
    }
    const int nb    = ntile * 16 + (lane & 15);     // D: lane&15 = N
    const int mhalf = (lane >> 4) * 8;              // D: vgpr r -> M = r + 8*half
    const float bv  = bias[nb];
#pragma unroll
    for (int m = 0; m < 6; ++m)
#pragma unroll
        for (int r = 0; r < 8; ++r) {
            int p = m * 16 + mhalf + r;
            if (p < 81) {
                float v = acc[m][r] + bv;
                if (res) v += (float)res[p * 128 + nb];
                v = v > 0.f ? v : 0.f;
                yout[p * 128 + nb] = (_Float16)v;
            }
        }
}

__global__ __launch_bounds__(256) void agent_forward(NetP P)
{
    const int agent = blockIdx.x;
    const int tid   = threadIdx.x;
    const int lane  = tid & 31;
    const int wave  = tid >> 5;        // ntile 0..7

    __shared__ __align__(16) _Float16 xA[FMAP_ROWS * 128];
    __shared__ __align__(16) _Float16 xB[FMAP_ROWS * 128];
    __shared__ float fsA[648];         // obs stage / conv2 flat / GRU gates
    __shared__ float fsB[384];
    __shared__ float fsC[256];         // latent[64] at [0..63]

    // ---- zero pad row (row 81) once; never written afterwards ----
    if (tid < 128) {
        xA[81 * 128 + tid] = (_Float16)0.f;
        xB[81 * 128 + tid] = (_Float16)0.f;
    }

    // ---- stage obs, conv1 (6->128, 3x3 pad1) via VALU (1% of FLOPs) ----
    const float* obsA = P.obs + (size_t)agent * 6 * 81;
    for (int i = tid; i < 486; i += 256) fsA[i] = obsA[i];
    __syncthreads();
    for (int idx = tid; idx < 81 * 128; idx += 256) {
        int co = idx & 127, p = idx >> 7;
        int h = p / 9, w = p % 9;
        float a = P.c1b[co];
        for (int ci = 0; ci < 6; ++ci)
            for (int t9 = 0; t9 < 9; ++t9) {
                int hh = h + t9 / 3 - 1, ww = w + t9 % 3 - 1;
                if ((unsigned)hh < 9u && (unsigned)ww < 9u)
                    a += fsA[ci * 81 + hh * 9 + ww] * P.c1w[(co * 6 + ci) * 9 + t9];
            }
        xA[idx] = (_Float16)(a > 0.f ? a : 0.f);
    }
    __syncthreads();

    // ---- residual conv blocks via WMMA ----
    conv_wmma(xA, xB, nullptr, P.wpack + 0 * 147456, P.r1ab, lane, wave);
    __syncthreads();
    conv_wmma(xB, xA, xA,      P.wpack + 1 * 147456, P.r1bb, lane, wave);
    __syncthreads();
    conv_wmma(xA, xB, nullptr, P.wpack + 2 * 147456, P.r2ab, lane, wave);
    __syncthreads();
    conv_wmma(xB, xA, xA,      P.wpack + 3 * 147456, P.r2bb, lane, wave);
    __syncthreads();

    // ---- conv2 1x1 (128->8) + relu, flatten as [o*81+p] ----
    for (int idx = tid; idx < 648; idx += 256) {
        int o = idx / 81, p = idx % 81;
        float a = P.c2b[o];
        for (int ci = 0; ci < 128; ++ci)
            a += (float)xA[p * 128 + ci] * P.c2w[o * 128 + ci];
        fsA[idx] = a > 0.f ? a : 0.f;
    }
    __syncthreads();

    // ---- fc_obs: 648 -> 48 ----
    if (tid < 48) {
        float a = P.fcb[tid];
        for (int f = 0; f < 648; ++f) a += fsA[f] * P.fcw[f * 48 + tid];
        fsC[tid] = a > 0.f ? a : 0.f;
    }
    // ---- pos branch: 4 -> 16, residual 16 ----
    if (tid < 16) {
        float a = P.pb[tid];
        for (int i2 = 0; i2 < 4; ++i2) a += P.pos[agent * 4 + i2] * P.pw[i2 * 16 + tid];
        fsB[tid] = a > 0.f ? a : 0.f;                       // pl
    }
    __syncthreads();
    if (tid < 16) {
        float a = P.p1b[tid];
        for (int i2 = 0; i2 < 16; ++i2) a += fsB[i2] * P.p1w[i2 * 16 + tid];
        fsB[32 + tid] = a > 0.f ? a : 0.f;                  // y1
    }
    __syncthreads();
    if (tid < 16) {
        float a = P.p2b[tid] + fsB[tid];
        for (int i2 = 0; i2 < 16; ++i2) a += fsB[32 + i2] * P.p2w[i2 * 16 + tid];
        fsC[48 + tid] = a > 0.f ? a : 0.f;                  // pos_lat
    }
    __syncthreads();

    // ---- cat residual MLP (64 -> 64, x2) ----
    if (tid < 64) {
        float a = P.cb1[tid];
        for (int i2 = 0; i2 < 64; ++i2) a += fsC[i2] * P.cw1[i2 * 64 + tid];
        fsB[tid] = a > 0.f ? a : 0.f;                       // t1
    }
    __syncthreads();
    if (tid < 64) {
        float a = P.cb2[tid] + fsC[tid];
        for (int i2 = 0; i2 < 64; ++i2) a += fsB[i2] * P.cw2[i2 * 64 + tid];
        fsB[64 + tid] = a > 0.f ? a : 0.f;                  // lat2
    }
    if (tid < 64) fsB[128 + tid] = P.hid[(size_t)agent * 64 + tid]; // h_old
    __syncthreads();

    // ---- GRU cell ----
    if (tid < 192) {
        float gi = P.bih[tid], gh = P.bhh[tid];
        for (int i2 = 0; i2 < 64; ++i2) {
            gi += fsB[64 + i2]  * P.wih[i2 * 192 + tid];
            gh += fsB[128 + i2] * P.whh[i2 * 192 + tid];
        }
        fsA[tid] = gi; fsA[192 + tid] = gh;
    }
    __syncthreads();
    if (tid < 64) {
        float r = 1.f / (1.f + expf(-(fsA[tid]       + fsA[192 + tid])));
        float z = 1.f / (1.f + expf(-(fsA[64 + tid]  + fsA[256 + tid])));
        float n = tanhf(fsA[128 + tid] + r * fsA[320 + tid]);
        P.hout[(size_t)agent * 64 + tid] = (1.f - z) * n + z * fsB[128 + tid];
    }
}

// ---------------------------------------------------------------------------
// Mask: argsort(d,row)<3 marks columns at the stable ranks of elements 0,1,2.
// Compute those 3 ranks per row, AND with in_obs (no abs in reference!).
// ---------------------------------------------------------------------------
__global__ __launch_bounds__(128) void comm_mask_kernel(
    const float* __restrict__ pos, int* __restrict__ tgt, int* __restrict__ rowcnt)
{
    const int i = blockIdx.x;
    const int t = threadIdx.x;
    const float xi = pos[i * 4 + 0], yi = pos[i * 4 + 1];
    __shared__ float dk[3];
    __shared__ int cnt[3];
    if (t < 3) {
        float dx = xi - pos[t * 4 + 0], dy = yi - pos[t * 4 + 1];
        dk[t] = sqrtf(dx * dx + dy * dy);
        cnt[t] = 0;
    }
    __syncthreads();
    const float d0 = dk[0], d1 = dk[1], d2 = dk[2];
    int c0 = 0, c1 = 0, c2 = 0;
    for (int j = t; j < N_AGENTS; j += 128) {
        float dx = xi - pos[j * 4 + 0], dy = yi - pos[j * 4 + 1];
        float d = sqrtf(dx * dx + dy * dy);
        c0 += (d < d0) ? 1 : 0;
        c1 += ((d < d1) || (d == d1 && j < 1)) ? 1 : 0;   // stable tie-break
        c2 += ((d < d2) || (d == d2 && j < 2)) ? 1 : 0;
    }
    atomicAdd(&cnt[0], c0); atomicAdd(&cnt[1], c1); atomicAdd(&cnt[2], c2);
    __syncthreads();
    if (t == 0) {
        int n = 0, tg[3] = {-1, -1, -1};
        for (int k = 0; k < 3; ++k) {
            int j = cnt[k];                                // rank of column k
            float dxo = xi - pos[j * 4 + 0], dyo = yi - pos[j * 4 + 1];
            if (dxo <= 4.0f && dyo <= 4.0f) tg[n++] = j;   // in_obs (asymmetric)
        }
        tgt[i * 3 + 0] = tg[0]; tgt[i * 3 + 1] = tg[1]; tgt[i * 3 + 2] = tg[2];
        rowcnt[i] = n;
    }
}

__global__ __launch_bounds__(256) void reduce_flag(
    const int* __restrict__ rowcnt, int* __restrict__ flag)
{
    __shared__ int s[256];
    int t = threadIdx.x, a = 0;
    for (int j = t; j < N_AGENTS; j += 256) a += rowcnt[j];
    s[t] = a; __syncthreads();
    for (int o = 128; o > 0; o >>= 1) { if (t < o) s[t] += s[t + o]; __syncthreads(); }
    if (t == 0) flag[0] = (s[0] <= 1) ? 1 : 0;
}

// ---------------------------------------------------------------------------
// One MHA layer with <=3 gathered neighbors per query. 64 threads per agent.
// ---------------------------------------------------------------------------
__global__ __launch_bounds__(64) void attn_layer(
    const float* __restrict__ hin, float* __restrict__ hout,
    const int* __restrict__ tgt, const int* __restrict__ flag,
    const float* __restrict__ wq, const float* __restrict__ bq,
    const float* __restrict__ wk, const float* __restrict__ bk,
    const float* __restrict__ wv, const float* __restrict__ bv,
    const float* __restrict__ wo, const float* __restrict__ bo)
{
    const int i = blockIdx.x;
    const int t = threadIdx.x;
    __shared__ float sh[64], sq[64], shj[64], sk[3][64], sv[3][64], sout[64];
    __shared__ float ssc[2][3], sp[2][3];
    __shared__ int stg[3];

    if (t < 3) stg[t] = tgt[i * 3 + t];
    if (t < 6) sp[t & 1][t >> 1] = 0.0f;
    sh[t] = hin[(size_t)i * 64 + t];
    __syncthreads();

    { float a = bq[t]; for (int u = 0; u < 64; ++u) a += sh[u] * wq[u * 64 + t]; sq[t] = a; }

    for (int s = 0; s < 3; ++s) {
        int j = stg[s];
        if (j >= 0) shj[t] = hin[(size_t)j * 64 + t];
        __syncthreads();
        if (j >= 0) {
            float ak = bk[t], av = bv[t];
            for (int u = 0; u < 64; ++u) {
                float hv = shj[u];
                ak += hv * wk[u * 64 + t];
                av += hv * wv[u * 64 + t];
            }
            sk[s][t] = ak; sv[s][t] = av;
        }
        __syncthreads();
    }

    if (t < 6) {
        int s = t >> 1, hd = t & 1;
        if (stg[s] >= 0) {
            float a = 0.f;
            for (int d = 0; d < 32; ++d) a += sq[hd * 32 + d] * sk[s][hd * 32 + d];
            ssc[hd][s] = a * 0.17677669529663687f;      // 1/sqrt(32)
        }
    }
    __syncthreads();

    if (t < 2) {
        int hd = t; float m = -3.0e38f; int any = 0;
        for (int s = 0; s < 3; ++s)
            if (stg[s] >= 0) { any = 1; if (ssc[hd][s] > m) m = ssc[hd][s]; }
        if (any) {
            float e[3] = {0.f, 0.f, 0.f}, sum = 0.f;
            for (int s = 0; s < 3; ++s)
                if (stg[s] >= 0) { e[s] = expf(ssc[hd][s] - m); sum += e[s]; }
            float inv = 1.f / sum;
            for (int s = 0; s < 3; ++s) sp[hd][s] = e[s] * inv;
        }
    }
    __syncthreads();

    { int hd = t >> 5; float a = 0.f;
      for (int s = 0; s < 3; ++s) if (stg[s] >= 0) a += sp[hd][s] * sv[s][t];
      sout[t] = a; }
    __syncthreads();

    float r = bo[t];
    for (int u = 0; u < 64; ++u) r += sout[u] * wo[u * 64 + t];
    hout[(size_t)i * 64 + t] = sh[t] + (flag[0] ? 0.f : r);
}

// ---------------------------------------------------------------------------
// Dueling head: q = st + adv - mean(adv)
// ---------------------------------------------------------------------------
__global__ __launch_bounds__(256) void head_kernel(
    const float* __restrict__ h,
    const float* __restrict__ advw, const float* __restrict__ advb,
    const float* __restrict__ stw,  const float* __restrict__ stb,
    float* __restrict__ out)
{
    int i = blockIdx.x * blockDim.x + threadIdx.x;
    if (i >= N_AGENTS) return;
    float adv[5]; for (int a = 0; a < 5; ++a) adv[a] = advb[a];
    float st = stb[0];
    for (int u = 0; u < 64; ++u) {
        float hv = h[(size_t)i * 64 + u];
        st += hv * stw[u];
        for (int a = 0; a < 5; ++a) adv[a] += hv * advw[u * 5 + a];
    }
    float m = 0.f; for (int a = 0; a < 5; ++a) m += adv[a];
    m *= 0.2f;
    for (int a = 0; a < 5; ++a) out[i * 5 + a] = st + adv[a] - m;
}

// ---------------------------------------------------------------------------
extern "C" void kernel_launch(void* const* d_in, const int* in_sizes, int n_in,
                              void* d_out, int out_size, void* d_ws, size_t ws_size,
                              hipStream_t stream)
{
    (void)in_sizes; (void)n_in; (void)out_size; (void)ws_size;
    const float* obs   = (const float*)d_in[0];
    const float* pos   = (const float*)d_in[1];
    const float* hid   = (const float*)d_in[2];
    const float* c1w   = (const float*)d_in[3];
    const float* c1b   = (const float*)d_in[4];
    const float* r1aw  = (const float*)d_in[5];
    const float* r1ab  = (const float*)d_in[6];
    const float* r1bw  = (const float*)d_in[7];
    const float* r1bb  = (const float*)d_in[8];
    const float* r2aw  = (const float*)d_in[9];
    const float* r2ab  = (const float*)d_in[10];
    const float* r2bw  = (const float*)d_in[11];
    const float* r2bb  = (const float*)d_in[12];
    const float* c2w   = (const float*)d_in[13];
    const float* c2b   = (const float*)d_in[14];
    const float* fcw   = (const float*)d_in[15];
    const float* fcb   = (const float*)d_in[16];
    const float* pw    = (const float*)d_in[17];
    const float* pb    = (const float*)d_in[18];
    const float* p1w   = (const float*)d_in[19];
    const float* p1b   = (const float*)d_in[20];
    const float* p2w   = (const float*)d_in[21];
    const float* p2b   = (const float*)d_in[22];
    const float* cw1   = (const float*)d_in[23];
    const float* cb1   = (const float*)d_in[24];
    const float* cw2   = (const float*)d_in[25];
    const float* cb2   = (const float*)d_in[26];
    const float* wih   = (const float*)d_in[27];
    const float* whh   = (const float*)d_in[28];
    const float* bih   = (const float*)d_in[29];
    const float* bhh   = (const float*)d_in[30];
    // mha params: 31..38 (layer0), 39..46 (layer1); head: 47..50
    const float* advw  = (const float*)d_in[47];
    const float* advb  = (const float*)d_in[48];
    const float* stw   = (const float*)d_in[49];
    const float* stb   = (const float*)d_in[50];

    char* ws = (char*)d_ws;
    _Float16* wpack = (_Float16*)(ws);                     // 1,179,648 B
    float* hbuf     = (float*)(ws + 1179648);              // 1,048,576 B
    float* hbuf2    = (float*)(ws + 1179648 + 1048576);    // 1,048,576 B
    int*   tgt      = (int*)(ws + 1179648 + 2 * 1048576);  //    49,152 B
    int*   rowcnt   = (int*)((char*)tgt + 49152);          //    16,384 B
    int*   flag     = (int*)((char*)rowcnt + 16384);       //         4 B

    pack_weights<<<(589824 + 255) / 256, 256, 0, stream>>>(r1aw, r1bw, r2aw, r2bw, wpack);

    NetP P;
    P.obs = obs; P.pos = pos; P.hid = hid;
    P.c1w = c1w; P.c1b = c1b;
    P.r1ab = r1ab; P.r1bb = r1bb; P.r2ab = r2ab; P.r2bb = r2bb;
    P.c2w = c2w; P.c2b = c2b; P.fcw = fcw; P.fcb = fcb;
    P.pw = pw; P.pb = pb; P.p1w = p1w; P.p1b = p1b; P.p2w = p2w; P.p2b = p2b;
    P.cw1 = cw1; P.cb1 = cb1; P.cw2 = cw2; P.cb2 = cb2;
    P.wih = wih; P.whh = whh; P.bih = bih; P.bhh = bhh;
    P.wpack = wpack; P.hout = hbuf;
    agent_forward<<<N_AGENTS, 256, 0, stream>>>(P);

    comm_mask_kernel<<<N_AGENTS, 128, 0, stream>>>(pos, tgt, rowcnt);
    reduce_flag<<<1, 256, 0, stream>>>(rowcnt, flag);

    attn_layer<<<N_AGENTS, 64, 0, stream>>>(hbuf, hbuf2, tgt, flag,
        (const float*)d_in[31], (const float*)d_in[32], (const float*)d_in[33],
        (const float*)d_in[34], (const float*)d_in[35], (const float*)d_in[36],
        (const float*)d_in[37], (const float*)d_in[38]);
    attn_layer<<<N_AGENTS, 64, 0, stream>>>(hbuf2, hbuf, tgt, flag,
        (const float*)d_in[39], (const float*)d_in[40], (const float*)d_in[41],
        (const float*)d_in[42], (const float*)d_in[43], (const float*)d_in[44],
        (const float*)d_in[45], (const float*)d_in[46]);

    head_kernel<<<N_AGENTS / 256, 256, 0, stream>>>(hbuf, advw, advb, stw, stb,
                                                    (float*)d_out);
}
